// RePN_1864015806994
// MI455X (gfx1250) — compile-verified, matrix operand order checked by reading, not run on hardware
//
#include <hip/hip_runtime.h>
#include <hip/hip_bf16.h>

typedef __attribute__((ext_vector_type(16))) __bf16 v16bf;
typedef __attribute__((ext_vector_type(8)))  __bf16 v8bf;
typedef __attribute__((ext_vector_type(8)))  float  v8f;
typedef unsigned u32x4 __attribute__((ext_vector_type(4)));
typedef unsigned u32x8 __attribute__((ext_vector_type(8)));

#define DI __device__ __forceinline__

constexpr int BATCH = 8;
constexpr int NPTS  = 2048;
constexpr int NC    = 150;
constexpr int NCP   = 160;   // K padded to 5*32
constexpr int HID   = 256;
constexpr int PROJ  = 1024;
constexpr int TOPK  = 128;
constexpr int ROWS  = BATCH * NPTS;          // 16384
constexpr int NNI   = NPTS * NPTS;           // 4,194,304

// ---- top-k state layout (unsigned words inside workspace) ----
constexpr int ST_HIST  = 0;      // 8*256
constexpr int ST_PRE   = 2048;   // 8
constexpr int ST_KREM  = 2056;   // 8
constexpr int ST_CNTG  = 2064;   // 8
constexpr int ST_CNTE  = 2072;   // 8
constexpr int ST_GIDX  = 2080;   // 8*128
constexpr int ST_EIDX  = 3104;   // 8*512
constexpr int ST_SEL   = 7200;   // 8*128
constexpr int ST_GKEY  = 8224;   // 8*128
constexpr int ST_WORDS = 9248;

// ---------------- WMMA fragment helpers ----------------
// 16-bit A-fragment (16x32, row-major source, lane L holds row L&15):
//   lanes 0-15 : elems 0..7 = K kb+0..7,  elems 8..15 = K kb+16..23
//   lanes 16-31: elems 0..7 = K kb+8..15, elems 8..15 = K kb+24..31
DI v16bf load_frag(const __bf16* base, int ld) {
  int lane = threadIdx.x & 31;
  const __bf16* p = base + (lane & 15) * ld + (lane >> 4) * 8;
  v8bf lo = *(const v8bf*)p;
  v8bf hi = *(const v8bf*)(p + 16);
  v16bf r;
#pragma unroll
  for (int i = 0; i < 8; ++i) { r[i] = lo[i]; r[i + 8] = hi[i]; }
  return r;
}

struct Acc4 { v8f c00, c01, c10, c11; };

DI void mm32x32(const __bf16* A, const __bf16* Bm, int lda, int ldb, int K, Acc4& acc) {
  for (int kb = 0; kb < K; kb += 32) {
    if (kb + 32 < K) {
      __builtin_prefetch(A + (threadIdx.x & 15) * lda + kb + 32, 0, 0);
      __builtin_prefetch(Bm + (threadIdx.x & 15) * ldb + kb + 32, 0, 0);
    }
    v16bf a0 = load_frag(A + kb, lda);
    v16bf a1 = load_frag(A + 16 * lda + kb, lda);
    v16bf b0 = load_frag(Bm + kb, ldb);
    v16bf b1 = load_frag(Bm + 16 * ldb + kb, ldb);
    acc.c00 = __builtin_amdgcn_wmma_f32_16x16x32_bf16(false, a0, false, b0, (short)0, acc.c00, false, false);
    acc.c01 = __builtin_amdgcn_wmma_f32_16x16x32_bf16(false, a0, false, b1, (short)0, acc.c01, false, false);
    acc.c10 = __builtin_amdgcn_wmma_f32_16x16x32_bf16(false, a1, false, b0, (short)0, acc.c10, false, false);
    acc.c11 = __builtin_amdgcn_wmma_f32_16x16x32_bf16(false, a1, false, b1, (short)0, acc.c11, false, false);
  }
}

// ---------------- TDM: 64x32 bf16 tile from (2048x1024) row-major -> LDS ----
// D# group0: count=1, lds_addr, 57-bit global tile addr, type=2.
// D# group1: data_size=1 (2B), tensor_dim0=1024, tensor_dim1=2048,
//            tile_dim0=32, tile_dim1=64, dim0_stride=1024 (elements).
DI void tdm_load_64x32(unsigned lds_addr, const __bf16* gptr) {
  unsigned long long ga = (unsigned long long)(uintptr_t)gptr;
  u32x4 g0;
  g0[0] = 1u;                                   // count=1, user descriptor
  g0[1] = lds_addr;                             // lds_addr bytes
  g0[2] = (unsigned)ga;                         // global_addr[31:0]
  g0[3] = (unsigned)(ga >> 32) | (2u << 30);    // global_addr[56:32] | type=2
  u32x8 g1;
  g1[0] = 0x00010000u;                          // data_size=1 (2 bytes)
  g1[1] = (unsigned)(PROJ & 0xFFFF) << 16;      // tensor_dim0 lo16 (=1024)
  g1[2] = (unsigned)(NPTS & 0xFFFF) << 16;      // dim0 hi16=0 | tensor_dim1 lo16 (=2048)
  g1[3] = 32u << 16;                            // dim1 hi16=0 | tile_dim0=32
  g1[4] = 64u;                                  // tile_dim1=64 | tile_dim2=0
  g1[5] = (unsigned)PROJ;                       // tensor_dim0_stride lo32 (=1024)
  g1[6] = 0u;                                   // stride hi | dim1_stride lo
  g1[7] = 0u;                                   // dim1_stride hi
  asm volatile("tensor_load_to_lds %0, %1" :: "s"(g0), "s"(g1) : "memory");
}

// ---------------- prep kernels ----------------
__global__ void k_cvt_scores(const float* __restrict__ sc, __bf16* __restrict__ dst) {
  int idx = blockIdx.x * 256 + threadIdx.x;
  if (idx >= ROWS * NCP) return;
  int r = idx / NCP, c = idx - r * NCP;
  float v = (c < NC) ? sc[r * (NC + 1) + c] : 0.f;
  dst[idx] = (__bf16)v;
}

__global__ void k_tc(const float* __restrict__ src, __bf16* __restrict__ dst,
                     int R, int C, int Kv, int sld) {
  int idx = blockIdx.x * 256 + threadIdx.x;
  if (idx >= R * C) return;
  int r = idx / C, c = idx - r * C;
  float v = (c < Kv) ? src[c * sld + r] : 0.f;
  dst[idx] = (__bf16)v;
}

// ---------------- GEMM1: H = relu(scB @ W1T^T + b1) ----------------
__global__ void k_gemm1(const __bf16* __restrict__ X, const __bf16* __restrict__ WT,
                        const float* __restrict__ bias, __bf16* __restrict__ H) {
  int t = blockIdx.x * 4 + (threadIdx.x >> 5);       // 4096 tiles: 512 x 8
  int tm = t >> 3, tn = t & 7;
  int rb = tm * 32, cb = tn * 32;
  Acc4 acc; acc.c00 = {}; acc.c01 = {}; acc.c10 = {}; acc.c11 = {};
  mm32x32(X + rb * NCP, WT + cb * NCP, NCP, NCP, NCP, acc);
  int lane = threadIdx.x & 31, lo = lane & 15, hi = lane >> 4;
  v8f cc[4] = {acc.c00, acc.c01, acc.c10, acc.c11};
#pragma unroll
  for (int s = 0; s < 4; ++s) {
    int dm = (s >> 1) * 16, dn = (s & 1) * 16;
#pragma unroll
    for (int r = 0; r < 8; ++r) {
      int m = rb + dm + r + hi * 8;
      int n = cb + dn + lo;
      float v = cc[s][r] + bias[n];
      v = v > 0.f ? v : 0.f;
      H[m * HID + n] = (__bf16)v;
    }
  }
}

// ---------------- GEMM2: Out = bf16((H @ W2T^T + b2) * feats) ----------------
__global__ void k_gemm2(const __bf16* __restrict__ H, const __bf16* __restrict__ WT,
                        const float* __restrict__ bias, const float* __restrict__ feats,
                        __bf16* __restrict__ Out) {
  int t = blockIdx.x * 4 + (threadIdx.x >> 5);       // 16384 tiles: 512 x 32
  int tm = t >> 5, tn = t & 31;
  int rb = tm * 32, cb = tn * 32;
  Acc4 acc; acc.c00 = {}; acc.c01 = {}; acc.c10 = {}; acc.c11 = {};
  mm32x32(H + rb * HID, WT + cb * HID, HID, HID, HID, acc);
  int lane = threadIdx.x & 31, lo = lane & 15, hi = lane >> 4;
  v8f cc[4] = {acc.c00, acc.c01, acc.c10, acc.c11};
#pragma unroll
  for (int s = 0; s < 4; ++s) {
    int dm = (s >> 1) * 16, dn = (s & 1) * 16;
#pragma unroll
    for (int r = 0; r < 8; ++r) {
      int m = rb + dm + r + hi * 8;
      int n = cb + dn + lo;
      float v = (cc[s][r] + bias[n]) * feats[m * PROJ + n];
      Out[m * PROJ + n] = (__bf16)v;
    }
  }
}

// ---------------- pairwise: 64x64 block tile, TDM-staged LDS, WMMA ----------
constexpr int CH = 64 * 32;                      // chunk elements (4 KB)

__global__ void k_pair(const __bf16* __restrict__ Aall, const __bf16* __restrict__ Ball,
                       float* __restrict__ out_sm) {
  __shared__ __align__(16) __bf16 bufA[2][CH];
  __shared__ __align__(16) __bf16 bufB[2][CH];
  int b = blockIdx.y;
  const __bf16* A  = Aall + (long long)b * NPTS * PROJ;
  const __bf16* Bm = Ball + (long long)b * NPTS * PROJ;
  float* outp = out_sm + (long long)b * NNI;
  int tmb = blockIdx.x >> 5, tnb = blockIdx.x & 31;   // 32x32 blocks of 64x64
  int ib = tmb * 64, jb = tnb * 64;
  int wave = threadIdx.x >> 5;
  int lane = threadIdx.x & 31, lo = lane & 15, hi = lane >> 4;
  int rb = ib + (wave >> 1) * 32;                     // this wave's 32x32 subtile
  int cb = jb + (wave & 1) * 32;

  if (tmb > tnb) {  // strictly-lower block: constant 0.5, no GEMM
#pragma unroll
    for (int s = 0; s < 4; ++s) {
      int dm = (s >> 1) * 16, dn = (s & 1) * 16;
#pragma unroll
      for (int r = 0; r < 8; ++r)
        outp[(rb + dm + r + hi * 8) * NPTS + (cb + dn + lo)] = 0.5f;
    }
    return;
  }

  unsigned ldsA = (unsigned)(uintptr_t)&bufA[0][0];   // ISA: LDS addr = flat[31:0]
  unsigned ldsB = (unsigned)(uintptr_t)&bufB[0][0];
  bool issuer = (__builtin_amdgcn_readfirstlane((int)threadIdx.x) < 32);  // wave 0, scalar branch

  if (issuer) {                                       // prologue: chunk 0 -> buffer 0
    tdm_load_64x32(ldsA, A + ib * PROJ);
    tdm_load_64x32(ldsB, Bm + jb * PROJ);
  }

  Acc4 acc; acc.c00 = {}; acc.c01 = {}; acc.c10 = {}; acc.c11 = {};
  for (int step = 0; step < PROJ / 32; ++step) {
    int cur = step & 1, nxt = cur ^ 1;
    if (issuer) {
      if (step < PROJ / 32 - 1) {                     // issue next chunk, then wait cur
        int kb = (step + 1) * 32;
        tdm_load_64x32(ldsA + nxt * (CH * 2), A + ib * PROJ + kb);
        tdm_load_64x32(ldsB + nxt * (CH * 2), Bm + jb * PROJ + kb);
        __builtin_amdgcn_s_wait_tensorcnt(2);
      } else {
        __builtin_amdgcn_s_wait_tensorcnt(0);
      }
    }
    __syncthreads();                                  // cur buffers ready for all waves
    const __bf16* pa = &bufA[cur][0];
    const __bf16* pb = &bufB[cur][0];
    v16bf a0 = load_frag(pa + ((wave >> 1) * 32) * 32, 32);
    v16bf a1 = load_frag(pa + ((wave >> 1) * 32 + 16) * 32, 32);
    v16bf b0 = load_frag(pb + ((wave & 1) * 32) * 32, 32);
    v16bf b1 = load_frag(pb + ((wave & 1) * 32 + 16) * 32, 32);
    acc.c00 = __builtin_amdgcn_wmma_f32_16x16x32_bf16(false, a0, false, b0, (short)0, acc.c00, false, false);
    acc.c01 = __builtin_amdgcn_wmma_f32_16x16x32_bf16(false, a0, false, b1, (short)0, acc.c01, false, false);
    acc.c10 = __builtin_amdgcn_wmma_f32_16x16x32_bf16(false, a1, false, b0, (short)0, acc.c10, false, false);
    acc.c11 = __builtin_amdgcn_wmma_f32_16x16x32_bf16(false, a1, false, b1, (short)0, acc.c11, false, false);
    __syncthreads();                                  // cur may be overwritten next step
  }

  v8f cc[4] = {acc.c00, acc.c01, acc.c10, acc.c11};
  float nanv = __uint_as_float(0x7FC00000u);
#pragma unroll
  for (int s = 0; s < 4; ++s) {
    int dm = (s >> 1) * 16, dn = (s & 1) * 16;
#pragma unroll
    for (int r = 0; r < 8; ++r) {
      int i = rb + dm + r + hi * 8;
      int j = cb + dn + lo;
      float m  = cc[s][r];
      float sg = 1.f / (1.f + __expf(-m));
      float v  = (i < j) ? sg : (i == j ? nanv : 0.5f);
      outp[i * NPTS + j] = v;
    }
  }
}

// ---------------- top-k: radix select on positive-float bits ----------------
__global__ void k_tk_init(unsigned* st) {
  int i = blockIdx.x * 256 + threadIdx.x;
  if (i < ST_WORDS) st[i] = (i >= ST_KREM && i < ST_KREM + 8) ? (unsigned)TOPK : 0u;
}
__global__ void k_hist_zero(unsigned* st) {
  int i = blockIdx.x * 256 + threadIdx.x;
  if (i < 2048) st[i] = 0u;
}
__global__ void k_hist(const float* __restrict__ sm, unsigned* __restrict__ st, int pass) {
  __shared__ unsigned lh[256];
  int b = blockIdx.y;
  lh[threadIdx.x] = 0u;
  __syncthreads();
  unsigned pre = st[ST_PRE + b];
  const float* p = sm + (long long)b * NNI;
  for (int idx = blockIdx.x * 256 + threadIdx.x; idx < NNI; idx += gridDim.x * 256) {
    int i = idx >> 11, j = idx & 2047;
    if (i == j) continue;                       // diagonal excluded (-inf in sel)
    unsigned key = __float_as_uint(p[idx]);
    bool ok = (pass == 0) || ((key >> (32 - 8 * pass)) == pre);
    if (ok) atomicAdd(&lh[(key >> (24 - 8 * pass)) & 255u], 1u);
  }
  __syncthreads();
  atomicAdd(&st[ST_HIST + b * 256 + threadIdx.x], lh[threadIdx.x]);
}
__global__ void k_pick(unsigned* st) {
  int b = threadIdx.x;
  if (b >= BATCH) return;
  unsigned rem = st[ST_KREM + b];
  const unsigned* h = &st[ST_HIST + b * 256];
  int bin = 255;
  for (; bin > 0; --bin) {
    unsigned c = h[bin];
    if (c >= rem) break;
    rem -= c;
  }
  st[ST_PRE + b]  = (st[ST_PRE + b] << 8) | (unsigned)bin;
  st[ST_KREM + b] = rem;
}
__global__ void k_collect(const float* __restrict__ sm, unsigned* __restrict__ st) {
  int b = blockIdx.y;
  unsigned T = st[ST_PRE + b];
  const float* p = sm + (long long)b * NNI;
  for (int idx = blockIdx.x * 256 + threadIdx.x; idx < NNI; idx += gridDim.x * 256) {
    int i = idx >> 11, j = idx & 2047;
    if (i == j) continue;
    unsigned key = __float_as_uint(p[idx]);
    if (key > T) {
      unsigned pos = atomicAdd(&st[ST_CNTG + b], 1u);
      if (pos < 128u) { st[ST_GIDX + b * 128 + pos] = (unsigned)idx; st[ST_GKEY + b * 128 + pos] = key; }
    } else if (key == T) {
      unsigned pos = atomicAdd(&st[ST_CNTE + b], 1u);
      if (pos < 512u) st[ST_EIDX + b * 512 + pos] = (unsigned)idx;
    }
  }
}
__global__ void k_final(unsigned* __restrict__ st, float* __restrict__ out_idx) {
  __shared__ unsigned gk[128], gi[128], ei[512];
  int b = blockIdx.x, t = threadIdx.x;                 // 128 threads
  int G = (int)min(st[ST_CNTG + b], 128u);
  int E = (int)min(st[ST_CNTE + b], 512u);
  int need = TOPK - G; if (need < 0) need = 0;
  if (t < G) { gk[t] = st[ST_GKEY + b * 128 + t]; gi[t] = st[ST_GIDX + b * 128 + t]; }
  for (int e = t; e < E; e += 128) ei[e] = st[ST_EIDX + b * 512 + e];
  __syncthreads();
  if (t < G) {
    unsigned k = gk[t], fi = gi[t];
    int rank = 0;
    for (int j = 0; j < G; ++j) rank += (gk[j] > k) || (gk[j] == k && gi[j] < fi);
    st[ST_SEL + b * 128 + rank] = fi;
    out_idx[b * TOPK + rank] = (float)fi;
  }
  for (int e = t; e < E; e += 128) {
    unsigned fi = ei[e];
    int rank = 0;
    for (int j = 0; j < E; ++j) rank += (ei[j] < fi);
    if (rank < need) {
      st[ST_SEL + b * 128 + G + rank] = fi;
      out_idx[b * TOPK + G + rank] = (float)fi;
    }
  }
}
__global__ void k_gather(const unsigned* __restrict__ st, const float* __restrict__ boxes,
                         const float* __restrict__ feats, float* __restrict__ out_boxes,
                         float* __restrict__ out_feats) {
  int bi = blockIdx.x, b = bi >> 7, t = bi & 127;
  unsigned p = st[ST_SEL + b * 128 + t];
  if (p >= (unsigned)NNI) p = 0;                       // guard against poison
  int si = (int)(p >> 11), oi = (int)(p & 2047);
  if (threadIdx.x < 8) {
    int s = threadIdx.x >> 2, c = threadIdx.x & 3;
    int src = (s == 0) ? si : oi;
    out_boxes[((b * TOPK + t) * 2 + s) * 4 + c] = boxes[(b * NPTS + src) * 4 + c];
  }
  const float* fs = feats + (long long)(b * NPTS + si) * PROJ;
  const float* fo = feats + (long long)(b * NPTS + oi) * PROJ;
  float* ob = out_feats + (long long)(b * TOPK + t) * 3 * PROJ;
  for (int d = threadIdx.x; d < PROJ; d += blockDim.x) {
    float a = fs[d], c2 = fo[d];
    ob[d] = a; ob[PROJ + d] = c2; ob[2 * PROJ + d] = 0.5f * (a + c2);
  }
}

extern "C" void kernel_launch(void* const* d_in, const int* in_sizes, int n_in,
                              void* d_out, int out_size, void* d_ws, size_t ws_size,
                              hipStream_t stream) {
  const float* scores   = (const float*)d_in[0];
  const float* features = (const float*)d_in[1];
  const float* boxes    = (const float*)d_in[2];
  const float* w1s = (const float*)d_in[3];
  const float* b1s = (const float*)d_in[4];
  const float* w2s = (const float*)d_in[5];
  const float* b2s = (const float*)d_in[6];
  const float* w1o = (const float*)d_in[7];
  const float* b1o = (const float*)d_in[8];
  const float* w2o = (const float*)d_in[9];
  const float* b2o = (const float*)d_in[10];

  // workspace carve-up (~90.4 MB)
  __bf16* scB  = (__bf16*)d_ws;
  __bf16* W1sT = scB  + (size_t)ROWS * NCP;
  __bf16* W1oT = W1sT + (size_t)HID * NCP;
  __bf16* W2sT = W1oT + (size_t)HID * NCP;
  __bf16* W2oT = W2sT + (size_t)PROJ * HID;
  __bf16* Hs   = W2oT + (size_t)PROJ * HID;
  __bf16* Ho   = Hs   + (size_t)ROWS * HID;
  __bf16* Abf  = Ho   + (size_t)ROWS * HID;
  __bf16* Bbf  = Abf  + (size_t)ROWS * PROJ;
  unsigned* st = (unsigned*)(Bbf + (size_t)ROWS * PROJ);

  float* out_sm    = (float*)d_out;
  float* out_idx   = out_sm    + (size_t)BATCH * NNI;
  float* out_boxes = out_idx   + BATCH * TOPK;
  float* out_feats = out_boxes + BATCH * TOPK * 2 * 4;

  // prep: convert/pad scores, transpose+convert weights to bf16
  k_cvt_scores<<<(ROWS * NCP + 255) / 256, 256, 0, stream>>>(scores, scB);
  k_tc<<<(HID * NCP + 255) / 256, 256, 0, stream>>>(w1s, W1sT, HID, NCP, NC, HID);
  k_tc<<<(HID * NCP + 255) / 256, 256, 0, stream>>>(w1o, W1oT, HID, NCP, NC, HID);
  k_tc<<<(PROJ * HID + 255) / 256, 256, 0, stream>>>(w2s, W2sT, PROJ, HID, HID, PROJ);
  k_tc<<<(PROJ * HID + 255) / 256, 256, 0, stream>>>(w2o, W2oT, PROJ, HID, HID, PROJ);

  // MLPs via WMMA
  k_gemm1<<<1024, 128, 0, stream>>>(scB, W1sT, b1s, Hs);
  k_gemm1<<<1024, 128, 0, stream>>>(scB, W1oT, b1o, Ho);
  k_gemm2<<<4096, 128, 0, stream>>>(Hs, W2sT, b2s, features, Abf);
  k_gemm2<<<4096, 128, 0, stream>>>(Ho, W2oT, b2o, features, Bbf);

  // pairwise Gram matrix via TDM-staged LDS + WMMA; sigmoid + triangular mask
  k_pair<<<dim3(1024, BATCH), 128, 0, stream>>>(Abf, Bbf, out_sm);

  // exact top-128 per batch via 4-pass radix select on float bits
  k_tk_init<<<(ST_WORDS + 255) / 256, 256, 0, stream>>>(st);
  for (int pass = 0; pass < 4; ++pass) {
    if (pass > 0) k_hist_zero<<<8, 256, 0, stream>>>(st);
    k_hist<<<dim3(2048, BATCH), 256, 0, stream>>>(out_sm, st, pass);
    k_pick<<<1, 32, 0, stream>>>(st);
  }
  k_collect<<<dim3(2048, BATCH), 256, 0, stream>>>(out_sm, st);
  k_final<<<BATCH, 128, 0, stream>>>(st, out_idx);
  k_gather<<<BATCH * TOPK, 256, 0, stream>>>(st, boxes, features, out_boxes, out_feats);

  (void)in_sizes; (void)n_in; (void)out_size; (void)ws_size;
}
